// AdvancedPredictionFramework_24919400252001
// MI455X (gfx1250) — matrix operand order repaired and checked by reading, not compiled, not run on hardware
//
#include <hip/hip_runtime.h>
#include <math.h>

// ---------------------------------------------------------------------------
// CDNA5 / gfx1250 WMMA types (wave32)
// ---------------------------------------------------------------------------
typedef __attribute__((ext_vector_type(16))) _Float16 v16h;
typedef __attribute__((ext_vector_type(8)))  float    v8f;

#define CDIM 128
#define MR   32            // rows per block tile (2 x 16-row WMMA sub-tiles)

__device__ __forceinline__ float wave_sum32(float v) {
#pragma unroll
  for (int off = 16; off > 0; off >>= 1) v += __shfl_xor(v, off, 32);
  return v;
}

__device__ __forceinline__ float gelu_exact(float x) {
  return 0.5f * x * (1.0f + erff(x * 0.7071067811865475f));
}

// ---------------------------------------------------------------------------
// Generic fused WMMA GEMM, specialized on (DIN, DOUT) so the K loop fully
// unrolls into back-to-back v_wmma_f32_16x16x32_f16.
//   out = epilogue( sum_t gather_t(X_t) @ W_t + bias )
// Weights are pre-packed f16, transposed: Wt[dout][din] -> per-lane B
// fragment is one contiguous 32-byte load.
// Block: (DOUT/16) waves; each wave owns one 16-wide column strip and two
// 16-row sub-tiles (shared B fragment).
// Epilogue modes:
//   0: store raw        1: LN + GELU, store     2: LN, store
//   3: LN, then out[r][c] += scale[r]*val (ensemble accumulate)
//   4: LN + GELU, then per-row dot(v2) + v2b -> sigmoid -> out[r]  (edge MLP)
// ---------------------------------------------------------------------------
struct GemmArgs {
  const float* X0; const float* X1; const float* X2; const float* X3;
  const int*   I0; const int*   I1;            // optional row-gather (terms 0,1)
  const _Float16* W0; const _Float16* W1; const _Float16* W2; const _Float16* W3;
  const float* bias;
  const float* ln_g; const float* ln_b;
  const float* scale; int scale_stride; int scale_off;  // mode 3
  const float* v2; const float* v2b;                    // mode 4
  float* out;
  int nrows; int nterms; int mode;
};

template <int DIN, int DOUT>
__global__ __launch_bounds__(DOUT * 2) void gemm_wmma_kernel(GemmArgs g) {
  constexpr int NT     = DOUT * 2;       // threads per block
  constexpr int NW     = DOUT / 16;      // waves per block
  constexpr int ASTR   = DIN + 8;        // LDS A row stride (halfs); byte mult of 16
  constexpr int KSTEPS = DIN / 32;

  __shared__ __align__(32) _Float16 sA[MR * ASTR];
  __shared__ __align__(16) float    sO[MR * 256];

  const int tid     = threadIdx.x;
  const int wid     = tid >> 5;
  const int lane    = tid & 31;
  const int tile0   = blockIdx.x * MR;
  const int n0      = wid << 4;
  const int halfSel = lane >> 4;         // 0: lanes 0-15, 1: lanes 16-31
  const int l15     = lane & 15;
  const int col     = n0 + l15;

  const float*    Xs[4] = {g.X0, g.X1, g.X2, g.X3};
  const int*      Is[4] = {g.I0, g.I1, nullptr, nullptr};
  const _Float16* Ws[4] = {g.W0, g.W1, g.W2, g.W3};

  v8f acc[2];
  acc[0] = (v8f){0.f, 0.f, 0.f, 0.f, 0.f, 0.f, 0.f, 0.f};
  acc[1] = acc[0];

  const int kb_a = halfSel * 8;    // unused; A fragment also reads at kb_b (k = 16h + s)
  const int kb_b = halfSel * 16;   // B-fragment K base offset for this half

  for (int t = 0; t < g.nterms; ++t) {
    // ---- stage A tile (MR x DIN) into LDS as f16, with optional gather ----
    const float* X   = Xs[t];
    const int*   idx = Is[t];
    const _Float16* Wt = Ws[t] + (long)col * DIN;
    __builtin_prefetch(Wt, 0, 0);                       // global_prefetch_b8
    for (int i = tid; i < MR * DIN; i += NT) {
      int r  = i / DIN;
      int c  = i & (DIN - 1);
      int gr = tile0 + r;
      float v = 0.f;
      if (gr < g.nrows) {
        long row = idx ? (long)idx[gr] : (long)gr;
        v = X[row * (long)DIN + c];
      }
      sA[r * ASTR + c] = (_Float16)v;
    }
    __syncthreads();

    // ---- fully unrolled K loop: 16x16x32 f16 WMMA, fp32 accumulate ----
#pragma unroll
    for (int ks = 0; ks < KSTEPS; ++ks) {
      const int k0 = ks * 32;
      // B fragment: 16 contiguous halves from transposed/packed weights
      v16h bf = *(const v16h*)(Wt + k0 + kb_b);
#pragma unroll
      for (int st = 0; st < 2; ++st) {
        union { v16h v; unsigned u[8]; } af;
        const unsigned* rowp =
            (const unsigned*)(sA + (st * 16 + l15) * ASTR);
#pragma unroll
        for (int p = 0; p < 8; ++p) {
          int kpair = kb_b + 2 * p;
          af.u[p] = rowp[(k0 + kpair) >> 1];
        }
        acc[st] = __builtin_amdgcn_wmma_f32_16x16x32_f16(
            false, af.v, false, bf, (short)0, acc[st], false, false);
      }
    }
    __syncthreads();
  }

  // ---- bias ----
  const float bia = g.bias ? g.bias[col] : 0.f;
#pragma unroll
  for (int st = 0; st < 2; ++st)
#pragma unroll
    for (int r = 0; r < 8; ++r) acc[st][r] += bia;

  if (g.mode == 0) {
#pragma unroll
    for (int st = 0; st < 2; ++st)
#pragma unroll
      for (int r = 0; r < 8; ++r) {
        int m  = st * 16 + r + halfSel * 8;   // D layout: VGPR r -> row r (+8 hi)
        int gr = tile0 + m;
        if (gr < g.nrows) g.out[(long)gr * DOUT + col] = acc[st][r];
      }
    return;
  }

  // ---- LayerNorm-based epilogues: spill tile to LDS ----
#pragma unroll
  for (int st = 0; st < 2; ++st)
#pragma unroll
    for (int r = 0; r < 8; ++r) {
      int m = st * 16 + r + halfSel * 8;
      sO[m * 256 + col] = acc[st][r];
    }
  __syncthreads();

  for (int r = wid; r < MR; r += NW) {
    int gr = tile0 + r;
    if (gr >= g.nrows) continue;
    float* row = sO + r * 256;
    float s = 0.f;
    for (int c = lane; c < DOUT; c += 32) s += row[c];
    s = wave_sum32(s);
    float mean = s / (float)DOUT;
    float vv = 0.f;
    for (int c = lane; c < DOUT; c += 32) { float d = row[c] - mean; vv += d * d; }
    vv = wave_sum32(vv);
    float rstd = rsqrtf(vv / (float)DOUT + 1e-5f);

    float dsum = 0.f;
    for (int c = lane; c < DOUT; c += 32) {
      float val = (row[c] - mean) * rstd * g.ln_g[c] + g.ln_b[c];
      if (g.mode == 1 || g.mode == 4) val = gelu_exact(val);
      if (g.mode == 1 || g.mode == 2) {
        g.out[(long)gr * DOUT + c] = val;
      } else if (g.mode == 3) {
        float sc = g.scale[(long)gr * g.scale_stride + g.scale_off];
        g.out[(long)gr * DOUT + c] += sc * val;
      } else {  // mode 4: dot with v2
        dsum += val * g.v2[c];
      }
    }
    if (g.mode == 4) {
      dsum = wave_sum32(dsum);
      if (lane == 0) {
        float z = dsum + g.v2b[0];
        g.out[gr] = 1.0f / (1.0f + expf(-z));
      }
    }
  }
}

// ---------------------------------------------------------------------------
// Weight packing: W[din][dout] f32 (row stride = dout) -> Wt[dout][din] f16
// ---------------------------------------------------------------------------
__global__ void pack_w_kernel(const float* __restrict__ src, _Float16* __restrict__ dst,
                              int din, int dout) {
  int i = blockIdx.x * blockDim.x + threadIdx.x;
  if (i >= din * dout) return;
  int n = i / din;
  int k = i - n * din;
  dst[i] = (_Float16)src[(long)k * dout + n];
}

// ---------------------------------------------------------------------------
// Small kernels
// ---------------------------------------------------------------------------
__global__ void scale_init_kernel(float* out, const float* a, float alpha, long n) {
  long i = (long)blockIdx.x * blockDim.x + threadIdx.x;
  if (i < n) out[i] = a ? alpha * a[i] : 0.f;
}

__global__ void edge_deg_kernel(const int* __restrict__ src, const int* __restrict__ dst,
                                const float* __restrict__ w, float* deg, int E) {
  int e = blockIdx.x * blockDim.x + threadIdx.x;
  if (e >= E) return;
  int s = src[e];
  if (s == dst[e]) return;
  float v = w ? w[e] : 1.f;
  if (v != 0.f) atomicAdd(deg + s, v);
}

__global__ void edge_norm_kernel(const int* __restrict__ src, const int* __restrict__ dst,
                                 const float* __restrict__ w, const float* __restrict__ deg,
                                 float* norm, int E) {
  int e = blockIdx.x * blockDim.x + threadIdx.x;
  if (e >= E) return;
  int s = src[e], d = dst[e];
  if (s == d) { norm[e] = 0.f; return; }
  float ww = w ? w[e] : 1.f;
  float ds_ = deg[s], dd = deg[d];
  float is_ = ds_ > 0.f ? rsqrtf(ds_) : 0.f;
  float id_ = dd  > 0.f ? rsqrtf(dd)  : 0.f;
  norm[e] = -is_ * ww * id_;
}

// out[dst[e]] += scale*norm[e]*xin[src[e]], vectorized 4 cols/thread
__global__ void prop_kernel(const float* __restrict__ xin, const int* __restrict__ src,
                            const int* __restrict__ dst, const float* __restrict__ norm,
                            float* out, int E, int cfeat, int shift, float scale) {
  long gid = (long)blockIdx.x * blockDim.x + threadIdx.x;
  long e = gid >> shift;
  if (e >= E) return;
  float nv = norm[e] * scale;
  if (nv == 0.f) return;
  int c4 = ((int)(gid & ((1 << shift) - 1))) << 2;
  const float4 xv = *(const float4*)(xin + (long)src[e] * cfeat + c4);
  float* orow = out + (long)dst[e] * cfeat + c4;
  atomicAdd(orow + 0, nv * xv.x);
  atomicAdd(orow + 1, nv * xv.y);
  atomicAdd(orow + 2, nv * xv.z);
  atomicAdd(orow + 3, nv * xv.w);
}

// wave-per-node: attn logits over concat(f0,f1,f2) (384->3), softmax, combine
__global__ void attn_combine_kernel(const float* __restrict__ F0, const float* __restrict__ F1,
                                    const float* __restrict__ F2, const float* __restrict__ aw,
                                    const float* __restrict__ ab, float* enh, int n) {
  int wv = (blockIdx.x * blockDim.x + threadIdx.x) >> 5;
  int lane = threadIdx.x & 31;
  if (wv >= n) return;
  const float* f0 = F0 + (long)wv * CDIM;
  const float* f1 = F1 + (long)wv * CDIM;
  const float* f2 = F2 + (long)wv * CDIM;
  float l0 = 0.f, l1 = 0.f, l2 = 0.f;
  for (int c = lane; c < CDIM; c += 32) {
    float v0 = f0[c], v1 = f1[c], v2v = f2[c];
    const float* w0 = aw + (long)c * 3;
    const float* w1 = aw + (long)(CDIM + c) * 3;
    const float* w2 = aw + (long)(2 * CDIM + c) * 3;
    l0 += v0 * w0[0] + v1 * w1[0] + v2v * w2[0];
    l1 += v0 * w0[1] + v1 * w1[1] + v2v * w2[1];
    l2 += v0 * w0[2] + v1 * w1[2] + v2v * w2[2];
  }
  l0 = wave_sum32(l0) + ab[0];
  l1 = wave_sum32(l1) + ab[1];
  l2 = wave_sum32(l2) + ab[2];
  float m = fmaxf(l0, fmaxf(l1, l2));
  float e0 = expf(l0 - m), e1 = expf(l1 - m), e2 = expf(l2 - m);
  float inv = 1.f / (e0 + e1 + e2);
  float a0 = e0 * inv, a1 = e1 * inv, a2 = e2 * inv;
  for (int c = lane; c < CDIM; c += 32)
    enh[(long)wv * CDIM + c] = a0 * f0[c] + a1 * f1[c] + a2 * f2[c];
}

// wave-per-node: gating head 128->3 + softmax
__global__ void gate_softmax_kernel(const float* __restrict__ H, const float* __restrict__ w,
                                    const float* __restrict__ b, float* ew, int n) {
  int wv = (blockIdx.x * blockDim.x + threadIdx.x) >> 5;
  int lane = threadIdx.x & 31;
  if (wv >= n) return;
  const float* h = H + (long)wv * CDIM;
  float l0 = 0.f, l1 = 0.f, l2 = 0.f;
  for (int c = lane; c < CDIM; c += 32) {
    float v = h[c];
    l0 += v * w[c * 3 + 0]; l1 += v * w[c * 3 + 1]; l2 += v * w[c * 3 + 2];
  }
  l0 = wave_sum32(l0) + b[0];
  l1 = wave_sum32(l1) + b[1];
  l2 = wave_sum32(l2) + b[2];
  float m = fmaxf(l0, fmaxf(l1, l2));
  float e0 = expf(l0 - m), e1 = expf(l1 - m), e2 = expf(l2 - m);
  float inv = 1.f / (e0 + e1 + e2);
  if (lane == 0) {
    ew[wv * 3 + 0] = e0 * inv; ew[wv * 3 + 1] = e1 * inv; ew[wv * 3 + 2] = e2 * inv;
  }
}

// wave-per-node: final 128->1 projection
__global__ void out_kernel(const float* __restrict__ HP, const float* __restrict__ w,
                           const float* __restrict__ b, float* out, int n) {
  int wv = (blockIdx.x * blockDim.x + threadIdx.x) >> 5;
  int lane = threadIdx.x & 31;
  if (wv >= n) return;
  float s = 0.f;
  for (int c = lane; c < CDIM; c += 32) s += HP[(long)wv * CDIM + c] * w[c];
  s = wave_sum32(s);
  if (lane == 0) out[wv] = s + b[0];
}

// ---------------------------------------------------------------------------
// Host: parameter index maps (pytree flatten order ambiguous -> detect)
// ---------------------------------------------------------------------------
struct PIdx {
  int x, ei;
  int s1w, s1b, slng, slnb, s2w, s2b;
  int cw[3][4]; int cb[3];
  int attw, attb;
  int ec1w[3][3], ec1b[3], eln1g[3], eln1b[3];
  int ec2w[3][2], ec2b[3], eln2g[3], eln2b[3];
  int wg1w, wg1b, wglng, wglnb, wg2w, wg2b;
  int p1w, p1b, plng, plnb, p2w, p2b;
};

extern "C" void kernel_launch(void* const* d_in, const int* in_sizes, int n_in,
                              void* d_out, int out_size, void* d_ws, size_t ws_size,
                              hipStream_t stream) {
  (void)out_size; (void)ws_size;
  PIdx P;
  // sorted-key flatten puts x last (largest leaf); insertion-order puts x first.
  bool sorted_order = (n_in >= 2) && (in_sizes[n_in - 1] > in_sizes[0]);
  if (!sorted_order) {
    P.x = 0; P.ei = 1;
    P.s1w = 2; P.s1b = 3; P.slng = 4; P.slnb = 5; P.s2w = 6; P.s2b = 7;
    P.cw[0][0] = 8;  P.cw[0][1] = 9;  P.cb[0] = 10;
    P.cw[1][0] = 11; P.cw[1][1] = 12; P.cw[1][2] = 13; P.cb[1] = 14;
    P.cw[2][0] = 15; P.cw[2][1] = 16; P.cw[2][2] = 17; P.cw[2][3] = 18; P.cb[2] = 19;
    P.attw = 20; P.attb = 21;
    for (int e = 0; e < 3; ++e) {
      int b = 22 + e * 11;
      P.ec1w[e][0] = b + 0; P.ec1w[e][1] = b + 1; P.ec1w[e][2] = b + 2; P.ec1b[e] = b + 3;
      P.eln1g[e] = b + 4; P.eln1b[e] = b + 5;
      P.ec2w[e][0] = b + 6; P.ec2w[e][1] = b + 7; P.ec2b[e] = b + 8;
      P.eln2g[e] = b + 9; P.eln2b[e] = b + 10;
    }
    P.wg1w = 55; P.wg1b = 56; P.wglng = 57; P.wglnb = 58; P.wg2w = 59; P.wg2b = 60;
    P.p1w = 67; P.p1b = 68; P.plng = 69; P.plnb = 70; P.p2w = 71; P.p2b = 72;
  } else {
    P.ei = 0;
    for (int e = 0; e < 3; ++e) {
      int b = 1 + e * 11;
      P.ec1b[e] = b + 0; P.ec1w[e][0] = b + 1; P.ec1w[e][1] = b + 2; P.ec1w[e][2] = b + 3;
      P.ec2b[e] = b + 4; P.ec2w[e][0] = b + 5; P.ec2w[e][1] = b + 6;
      P.eln1b[e] = b + 7; P.eln1g[e] = b + 8; P.eln2b[e] = b + 9; P.eln2g[e] = b + 10;
    }
    P.wg1b = 40; P.wg1w = 41; P.wg2b = 42; P.wg2w = 43; P.wglnb = 44; P.wglng = 45;
    P.p1b = 46; P.p1w = 47; P.p2b = 48; P.p2w = 49; P.plnb = 50; P.plng = 51;
    P.attb = 52; P.attw = 53;
    P.cb[0] = 54; P.cw[0][0] = 55; P.cw[0][1] = 56;
    P.cb[1] = 57; P.cw[1][0] = 58; P.cw[1][1] = 59; P.cw[1][2] = 60;
    P.cb[2] = 61; P.cw[2][0] = 62; P.cw[2][1] = 63; P.cw[2][2] = 64; P.cw[2][3] = 65;
    P.s1b = 66; P.s1w = 67; P.s2b = 68; P.s2w = 69; P.slnb = 70; P.slng = 71;
    P.x = 72;
  }

  auto F = [&](int i) { return (const float*)d_in[i]; };
  const float* x = F(P.x);
  const int* ei  = (const int*)d_in[P.ei];
  const int E = in_sizes[P.ei] / 2;
  const int N = in_sizes[P.x] / CDIM;
  const int* src = ei;
  const int* dst = ei + E;

  // ---- workspace layout (floats), with aliasing reuse ----
  float* ws = (float*)d_ws;
  long off = 0;
  float* DEG   = ws + off; off += ((long)N + 7) / 8 * 8;
  float* SCORE = ws + off; off += ((long)E + 7) / 8 * 8;
  float* NORMS = ws + off; off += ((long)E + 7) / 8 * 8;
  float* NORM1 = ws + off; off += ((long)E + 7) / 8 * 8;
  float* EW    = ws + off; off += ((long)3 * N + 7) / 8 * 8;
  const long NC = (long)N * CDIM;
  float* B1  = ws + off;        // stage-1 base b1 ; later ensemble base eb1
  float* B2  = B1 + 1 * NC;     // b2 ; later eb2
  float* F0b = B1 + 2 * NC;     // feat0 ; later expert H1 (spans F0b,F1b)
  float* F1b = B1 + 3 * NC;     // feat1
  float* F2b = B1 + 4 * NC;     // feat2 ; later HB1 (spans F2b,B3b)
  float* B3b = B1 + 5 * NC;     // b3 ; later gating hidden WGH
  float* ENH = B1 + 6 * NC;
  float* ENS = B1 + 7 * NC;
  float* HP  = B1 + 8 * NC;
  float* H1  = F0b;
  float* HB1 = F2b;
  float* WGH = B3b;
  _Float16* hwcur = (_Float16*)(B1 + 9 * NC);   // packed f16 weight arena

  auto pack = [&](const float* srcw, int din, int dout) -> const _Float16* {
    _Float16* d = hwcur;
    long n = (long)din * dout;
    hwcur += n;
    pack_w_kernel<<<dim3((unsigned)((n + 255) / 256)), dim3(256), 0, stream>>>(
        srcw, d, din, dout);
    return d;
  };
  auto initbuf = [&](float* o, const float* a, float alpha, long n) {
    long blocks = (n + 255) / 256;
    scale_init_kernel<<<dim3((unsigned)blocks), dim3(256), 0, stream>>>(o, a, alpha, n);
  };
  auto run_prop = [&](const float* xin, const float* nrm, float* ob, int cfeat, float scale) {
    int shift = (cfeat == 256) ? 6 : 5;
    long total = (long)E << shift;
    long blocks = (total + 255) / 256;
    prop_kernel<<<dim3((unsigned)blocks), dim3(256), 0, stream>>>(
        xin, src, dst, nrm, ob, E, cfeat, shift, scale);
  };
  auto gemm = [&](int nrows, int din, int dout, int nterms,
                  const float* X0, const float* X1, const float* X2, const float* X3,
                  const int* I0, const int* I1,
                  const _Float16* W0, const _Float16* W1, const _Float16* W2,
                  const _Float16* W3,
                  const float* bias, int mode, const float* lg, const float* lb,
                  const float* scale, int ss, int so,
                  const float* v2, const float* v2b, float* outp) {
    GemmArgs ga;
    ga.X0 = X0; ga.X1 = X1; ga.X2 = X2; ga.X3 = X3;
    ga.I0 = I0; ga.I1 = I1;
    ga.W0 = W0; ga.W1 = W1; ga.W2 = W2; ga.W3 = W3;
    ga.bias = bias; ga.ln_g = lg; ga.ln_b = lb;
    ga.scale = scale; ga.scale_stride = ss; ga.scale_off = so;
    ga.v2 = v2; ga.v2b = v2b; ga.out = outp;
    ga.nrows = nrows; ga.nterms = nterms; ga.mode = mode;
    int blocks = (nrows + MR - 1) / MR;
    if (din == 128 && dout == 128)
      gemm_wmma_kernel<128, 128><<<dim3(blocks), dim3(256), 0, stream>>>(ga);
    else if (din == 128 && dout == 256)
      gemm_wmma_kernel<128, 256><<<dim3(blocks), dim3(512), 0, stream>>>(ga);
    else
      gemm_wmma_kernel<256, 128><<<dim3(blocks), dim3(256), 0, stream>>>(ga);
  };
  const int eb = (E + 255) / 256;
  const int nb = (N + 7) / 8;   // wave-per-node kernels, 8 nodes/block

  // ---- 1. edge MLP -> edge scores (fused gather-GEMM + LN + GELU + dot + sigmoid)
  {
    const _Float16* w0 = pack(F(P.s1w), 128, 128);
    const _Float16* w1 = pack(F(P.s1w) + CDIM * CDIM, 128, 128);
    gemm(E, 128, 128, 2, x, x, nullptr, nullptr, src, dst,
         w0, w1, nullptr, nullptr, F(P.s1b),
         4, F(P.slng), F(P.slnb), nullptr, 0, 0, F(P.s2w), F(P.s2b), SCORE);
  }

  // ---- 2. cheb_norm(score)
  initbuf(DEG, nullptr, 0.f, N);
  edge_deg_kernel<<<eb, 256, 0, stream>>>(src, dst, SCORE, DEG, E);
  edge_norm_kernel<<<eb, 256, 0, stream>>>(src, dst, SCORE, DEG, NORMS, E);

  // ---- 3. Chebyshev bases K=4 (b0 = x)
  initbuf(B1, nullptr, 0.f, NC);    run_prop(x,  NORMS, B1, 128, 1.f);
  initbuf(B2, x, -1.f, NC);         run_prop(B1, NORMS, B2, 128, 2.f);
  initbuf(B3b, B1, -1.f, NC);       run_prop(B2, NORMS, B3b, 128, 2.f);

  // ---- 4. feats (cheb_apply K=2,3,4)
  gemm(N, 128, 128, 2, x, B1, nullptr, nullptr, nullptr, nullptr,
       pack(F(P.cw[0][0]), 128, 128), pack(F(P.cw[0][1]), 128, 128), nullptr, nullptr,
       F(P.cb[0]), 0, nullptr, nullptr, nullptr, 0, 0, nullptr, nullptr, F0b);
  gemm(N, 128, 128, 3, x, B1, B2, nullptr, nullptr, nullptr,
       pack(F(P.cw[1][0]), 128, 128), pack(F(P.cw[1][1]), 128, 128),
       pack(F(P.cw[1][2]), 128, 128), nullptr,
       F(P.cb[1]), 0, nullptr, nullptr, nullptr, 0, 0, nullptr, nullptr, F1b);
  gemm(N, 128, 128, 4, x, B1, B2, B3b, nullptr, nullptr,
       pack(F(P.cw[2][0]), 128, 128), pack(F(P.cw[2][1]), 128, 128),
       pack(F(P.cw[2][2]), 128, 128), pack(F(P.cw[2][3]), 128, 128),
       F(P.cb[2]), 0, nullptr, nullptr, nullptr, 0, 0, nullptr, nullptr, F2b);

  // ---- 5. attention softmax + combine -> enhanced
  attn_combine_kernel<<<nb, 256, 0, stream>>>(F0b, F1b, F2b, F(P.attw), F(P.attb), ENH, N);

  // ---- 6. cheb_norm(ones)
  initbuf(DEG, nullptr, 0.f, N);
  edge_deg_kernel<<<eb, 256, 0, stream>>>(src, dst, nullptr, DEG, E);
  edge_norm_kernel<<<eb, 256, 0, stream>>>(src, dst, nullptr, DEG, NORM1, E);

  // ---- 7. ensemble bases K=3 on enhanced (eb0 = ENH; reuse B1,B2)
  initbuf(B1, nullptr, 0.f, NC);    run_prop(ENH, NORM1, B1, 128, 1.f);
  initbuf(B2, ENH, -1.f, NC);       run_prop(B1,  NORM1, B2, 128, 2.f);

  // ---- 8. gating weights (before experts so B3b/WGH frees for HB1)
  gemm(N, 128, 128, 1, ENH, nullptr, nullptr, nullptr, nullptr, nullptr,
       pack(F(P.wg1w), 128, 128), nullptr, nullptr, nullptr, F(P.wg1b),
       1, F(P.wglng), F(P.wglnb), nullptr, 0, 0, nullptr, nullptr, WGH);
  gate_softmax_kernel<<<nb, 256, 0, stream>>>(WGH, F(P.wg2w), F(P.wg2b), EW, N);

  // ---- 9. experts, accumulated into ENS weighted by EW
  initbuf(ENS, nullptr, 0.f, NC);
  for (int e = 0; e < 3; ++e) {
    gemm(N, 128, 256, 3, ENH, B1, B2, nullptr, nullptr, nullptr,
         pack(F(P.ec1w[e][0]), 128, 256), pack(F(P.ec1w[e][1]), 128, 256),
         pack(F(P.ec1w[e][2]), 128, 256), nullptr,
         F(P.ec1b[e]), 1, F(P.eln1g[e]), F(P.eln1b[e]),
         nullptr, 0, 0, nullptr, nullptr, H1);
    initbuf(HB1, nullptr, 0.f, 2 * NC);
    run_prop(H1, NORM1, HB1, 256, 1.f);
    gemm(N, 256, 128, 2, H1, HB1, nullptr, nullptr, nullptr, nullptr,
         pack(F(P.ec2w[e][0]), 256, 128), pack(F(P.ec2w[e][1]), 256, 128),
         nullptr, nullptr,
         F(P.ec2b[e]), 3, F(P.eln2g[e]), F(P.eln2b[e]),
         EW, 3, e, nullptr, nullptr, ENS);
  }

  // ---- 10. predictor
  gemm(N, 128, 128, 2, ENH, ENS, nullptr, nullptr, nullptr, nullptr,
       pack(F(P.p1w), 128, 128), pack(F(P.p1w) + CDIM * CDIM, 128, 128),
       nullptr, nullptr, F(P.p1b),
       1, F(P.plng), F(P.plnb), nullptr, 0, 0, nullptr, nullptr, HP);
  out_kernel<<<nb, 256, 0, stream>>>(HP, F(P.p2w), F(P.p2b), (float*)d_out, N);
}
